// SingleHeadAttention_74448963109518
// MI455X (gfx1250) — matrix operand order repaired and checked by reading, not compiled, not run on hardware
//
#include <hip/hip_runtime.h>

// ---------------------------------------------------------------------------
// Single-head causal attention for MI455X (gfx1250), wave32 + WMMA bf16 +
// Tensor Data Mover (async B-tile staging into LDS).
//
// Pipeline (all intermediates fit in the 192MB L2):
//   K0  convert x / w_qkv / w_out  fp32 -> bf16
//   K1  qkv = x @ w_qkv^T + b   -> Q[B,S,H], K[B,S,H], V^T[B,H,S]  (bf16)
//   K2  scores = Q @ K^T * (1/32), lower-triangular tiles only (fp32)
//   K3  causal softmax over k<=q  -> probs bf16 (zeros above diagonal)
//   K4  ao = probs @ V           (uses V^T so B-operand is K-contiguous)
//   K5  out = ao @ w_out^T + b   (fp32 output)
//
// GEMM block: 128(M) x 64(N) per workgroup (8 waves x 16 rows), K stage 64.
//  - B tile (shared by all 8 waves) staged in LDS, double buffered with
//    STATIC buffer pointers (pair-unrolled loop -> no accumulator phi-copies),
//    filled by TENSOR_LOAD_TO_LDS issued from wave 0. TDM pad feature gives a
//    160B LDS row stride: every fragment address is 32B aligned.
//  - A fragments register-pipelined one stage ahead straight from global.
//  - s_wait_tensorcnt executed wave-uniformly by all waves (no exec juggling).
// ---------------------------------------------------------------------------

#define B_ 4
#define S_ 2048
#define H_ 1024

#define LDSROW 80   // bf16 elements per LDS B-tile row (128B data + 32B pad)

typedef __bf16 bf16_t;
typedef __attribute__((ext_vector_type(16))) __bf16 v16bf;
typedef __attribute__((ext_vector_type(8)))  __bf16 v8bf;
typedef __attribute__((ext_vector_type(4)))  __bf16 v4bf;
typedef __attribute__((ext_vector_type(8)))  float  v8f;
typedef __attribute__((ext_vector_type(4)))  float  v4f;
typedef __attribute__((ext_vector_type(4)))  unsigned int v4u;
typedef __attribute__((ext_vector_type(8)))  int    v8i;
typedef __attribute__((ext_vector_type(4)))  int    v4i;

#if defined(__has_builtin)
#if __has_builtin(__builtin_amdgcn_tensor_load_to_lds) && \
    __has_builtin(__builtin_amdgcn_s_wait_tensorcnt)
#define HAVE_TDM 1
#else
#define HAVE_TDM 0
#endif
#else
#define HAVE_TDM 0
#endif

static __device__ __forceinline__ v8f wmma_bf16(v16bf a, v16bf b, v8f c) {
  // (neg_a, A, neg_b, B, c_mod, C, reuse_a, reuse_b)
  return __builtin_amdgcn_wmma_f32_16x16x32_bf16(false, a, false, b,
                                                 (short)0, c, false, false);
}

// A fragment, 16x32 bf16 (M x K). Lanes hold M=lane%16,
// contiguous K 16*half..16*half+15 (matches B fragment).
static __device__ __forceinline__ v16bf load_a_frag(const bf16_t* base, int lda,
                                                    int k0, int lane) {
  const int hf  = lane >> 4;
  const int row = lane & 15;
  const bf16_t* p = base + (size_t)row * lda + k0 + hf * 16;
  v8bf lo = *(const v8bf*)(p);
  v8bf hi = *(const v8bf*)(p + 8);
  return __builtin_shufflevector(lo, hi, 0, 1, 2, 3, 4, 5, 6, 7,
                                 8, 9, 10, 11, 12, 13, 14, 15);
}

// B fragment, 32x16 bf16 (K x N), from the LDS-staged stage tile
// (N-major rows of LDSROW elements, two 32-K slabs per row).
// Lane = column N, contiguous 16-element K run. 160B row stride keeps
// every address 32B aligned -> single v16bf load (2x ds_load_b128).
static __device__ __forceinline__ v16bf lds_b_frag(const bf16_t* lb, int t,
                                                   int slab, int lane) {
  const int hf  = lane >> 4;
  const int col = lane & 15;
  return *(const v16bf*)(lb + (t * 16 + col) * LDSROW + slab * 32 + hf * 16);
}

#if HAVE_TDM
// Issue one TDM 2D tile load: 64 rows x 128B (K-stage of the B operand) from
// global (row stride ldb bf16) into LDS with +32B pad per 128B row.
static __device__ __forceinline__ void tdm_issue_b_tile(const bf16_t* gsrc,
                                                        int ldb, int k0,
                                                        bf16_t* lds_dst) {
  const unsigned long long ga =
      (unsigned long long)(const char*)gsrc + (unsigned long long)k0 * 2ull;
  const unsigned lds_off = (unsigned)(unsigned long long)lds_dst; // LDS byte offset
  const unsigned row8 = (unsigned)ldb >> 2;  // row stride/length in 8B units

  v4u g0;
  g0[0] = 1u;                                 // count=1 (valid), user mode
  g0[1] = lds_off;                            // lds_addr
  g0[2] = (unsigned)ga;                       // global_addr[31:0]
  g0[3] = (unsigned)(ga >> 32) | (2u << 30);  // global_addr[56:32] | type=2

  v8i g1;
  g1[0] = (int)((3u << 16)     // data_size = 8B
              | (1u << 20)     // pad_enable
              | (4u << 22)     // pad_interval: 32 DWORDs (128B)
              | (7u << 25));   // pad_amount:   8 DWORDs (32B)
  g1[1] = (int)((row8 & 0xffffu) << 16);      // tensor_dim0[15:0]
  g1[2] = (int)((row8 >> 16) & 0xffffu);      // tensor_dim0[31:16] | dim1 lo=0
  g1[3] = (int)(1u | (16u << 16));            // tensor_dim1=65536 | tile_dim0=16
  g1[4] = 64;                                 // tile_dim1 = 64 rows (tile_dim2=0)
  g1[5] = (int)row8;                          // tensor_dim0_stride low 32
  g1[6] = 0;
  g1[7] = 0;

  v4i g2; g2[0] = 0; g2[1] = 0; g2[2] = 0; g2[3] = 0;
  v4i g3; g3[0] = 0; g3[1] = 0; g3[2] = 0; g3[3] = 0;

#if __clang_major__ >= 23
  v8i g4; g4[0]=0; g4[1]=0; g4[2]=0; g4[3]=0; g4[4]=0; g4[5]=0; g4[6]=0; g4[7]=0;
  __builtin_amdgcn_tensor_load_to_lds(g0, g1, g2, g3, g4, 0);
#else
  __builtin_amdgcn_tensor_load_to_lds(g0, g1, g2, g3, 0);
#endif
}
#endif  // HAVE_TDM

// Start filling one LDS B-tile buffer with the 64-wide K-stage at k0.
static __device__ __forceinline__ void b_tile_fill(bf16_t* dst,
                                                   const bf16_t* Bblock,
                                                   int ldb, int k0, int tid) {
#if HAVE_TDM
  if (tid < 32) tdm_issue_b_tile(Bblock, ldb, k0, dst);  // wave 0 drives TDM
#else
  for (int c = tid; c < 1024; c += 256) {  // 64 rows x 16 chunks of 8B
    const int row = c >> 4, ch = c & 15;
    *(unsigned long long*)(dst + row * LDSROW + ch * 4) =
        *(const unsigned long long*)(Bblock + (size_t)row * ldb + k0 + ch * 4);
  }
#endif
}

static __device__ __forceinline__ void b_tile_wait() {
#if HAVE_TDM
  // Scalar wait, wave-uniform: waves that issued nothing pass immediately.
  __builtin_amdgcn_s_wait_tensorcnt(0);
#endif
}

// One K-stage (K=64): 8 back-to-back WMMAs; all B-frag ds loads of a slab
// grouped ahead of the WMMA group so the ds waits coalesce.
static __device__ __forceinline__ void stage8(const bf16_t* buf, v16bf a0,
                                              v16bf a1, int lane, v8f acc[4]) {
  v16bf b0[4], b1[4];
#pragma unroll
  for (int t = 0; t < 4; ++t) b0[t] = lds_b_frag(buf, t, 0, lane);
#pragma unroll
  for (int t = 0; t < 4; ++t) acc[t] = wmma_bf16(a0, b0[t], acc[t]);
#pragma unroll
  for (int t = 0; t < 4; ++t) b1[t] = lds_b_frag(buf, t, 1, lane);
#pragma unroll
  for (int t = 0; t < 4; ++t) acc[t] = wmma_bf16(a1, b1[t], acc[t]);
}

// Workgroup GEMM: per-wave 16(M) x 64(N) accumulators; B tile double buffered
// in LDS with static buffer pointers (pair-unrolled), TDM fill and A-fragment
// register prefetch overlapped with the WMMA stage. K multiple of 128.
static __device__ __forceinline__ void gemm_block(const bf16_t* Awave, int lda,
                                                  const bf16_t* Bblock, int ldb,
                                                  int K, int tid,
                                                  bf16_t* lds0, bf16_t* lds1,
                                                  v8f acc[4]) {
  const int lane = tid & 31;
#pragma unroll
  for (int t = 0; t < 4; ++t)
#pragma unroll
    for (int j = 0; j < 8; ++j) acc[t][j] = 0.0f;

  b_tile_fill(lds0, Bblock, ldb, 0, tid);
  b_tile_wait();
  __syncthreads();

  v16bf a0 = load_a_frag(Awave, lda, 0, lane);
  v16bf a1 = load_a_frag(Awave, lda, 32, lane);

  const int NS = K >> 6;  // 64-wide stages; NS is even for K in {1024, 2048}
  for (int s = 0; s < NS; s += 2) {
    const int kb1 = (s + 1) << 6;
    const int kb2 = (s + 2) << 6;

    // ---- stage s: compute from lds0, fill lds1, prefetch A for stage s+1
    if (s + 1 < NS) b_tile_fill(lds1, Bblock, ldb, kb1, tid);
    v16bf c0 = a0, c1 = a1;
    if (s + 1 < NS) {
      a0 = load_a_frag(Awave, lda, kb1, lane);
      a1 = load_a_frag(Awave, lda, kb1 + 32, lane);
    }
    stage8(lds0, c0, c1, lane, acc);
    b_tile_wait();
    __syncthreads();
    if (s + 1 >= NS) break;

    // ---- stage s+1: compute from lds1, fill lds0, prefetch A for stage s+2
    if (s + 2 < NS) {
      b_tile_fill(lds0, Bblock, ldb, kb2, tid);
      __builtin_prefetch(Awave + (size_t)(lane & 15) * lda + kb2, 0, 1);
    }
    c0 = a0; c1 = a1;
    if (s + 2 < NS) {
      a0 = load_a_frag(Awave, lda, kb2, lane);
      a1 = load_a_frag(Awave, lda, kb2 + 32, lane);
    }
    stage8(lds1, c0, c1, lane, acc);
    b_tile_wait();
    __syncthreads();
  }
}

// ---------------------------------------------------------------------------
// K0: fp32 -> bf16 convert (vectorized x4)
// ---------------------------------------------------------------------------
__global__ void __launch_bounds__(256)
cvt_f32_to_bf16(const float* __restrict__ in, bf16_t* __restrict__ out, int n4) {
  int i = blockIdx.x * 256 + threadIdx.x;
  if (i >= n4) return;
  v4f x = ((const v4f*)in)[i];
  v4bf o;
#pragma unroll
  for (int j = 0; j < 4; ++j) o[j] = (bf16_t)x[j];
  ((v4bf*)out)[i] = o;
}

// ---------------------------------------------------------------------------
// K1: qkv projection. Block tile 128(M) x 64(N), M = flattened B*S rows.
//     N in [0,H) -> Q, [H,2H) -> K, [2H,3H) -> V stored transposed.
// ---------------------------------------------------------------------------
__global__ void __launch_bounds__(256)
qkv_gemm(const bf16_t* __restrict__ xb, const bf16_t* __restrict__ wb,
         const float* __restrict__ bias,
         bf16_t* __restrict__ q, bf16_t* __restrict__ k, bf16_t* __restrict__ vT) {
  __shared__ bf16_t ldsB[2][64 * LDSROW];
  const int tid = threadIdx.x;
  const int lane = tid & 31;
  const int wave = tid >> 5;
  const int m0 = blockIdx.x * 128 + wave * 16;
  const int n0 = blockIdx.y * 64;

  v8f acc[4];
  gemm_block(xb + (size_t)m0 * H_, H_, wb + (size_t)n0 * H_, H_, H_, tid,
             &ldsB[0][0], &ldsB[1][0], acc);

  const int hf = lane >> 4, col = lane & 15;
#pragma unroll
  for (int t = 0; t < 4; ++t) {
    const int n = n0 + t * 16 + col;
    const float bv = bias[n];
    if (n < H_) {
#pragma unroll
      for (int j = 0; j < 8; ++j) {
        const int m = m0 + hf * 8 + j;
        q[(size_t)m * H_ + n] = (bf16_t)(acc[t][j] + bv);
      }
    } else if (n < 2 * H_) {
#pragma unroll
      for (int j = 0; j < 8; ++j) {
        const int m = m0 + hf * 8 + j;
        k[(size_t)m * H_ + (n - H_)] = (bf16_t)(acc[t][j] + bv);
      }
    } else {
      // V^T[b][h][s]: lane holds one h (=n-2H) and 8 consecutive s -> one
      // contiguous 16B store.
      const int h = n - 2 * H_;
      const int m = m0 + hf * 8;       // 16-row tile never crosses a batch
      const int b = m / S_;
      const int s = m % S_;
      v8bf pk;
#pragma unroll
      for (int j = 0; j < 8; ++j) pk[j] = (bf16_t)(acc[t][j] + bv);
      *(v8bf*)(vT + ((size_t)b * H_ + h) * S_ + s) = pk;
    }
  }
}

// ---------------------------------------------------------------------------
// K2: scores = Q @ K^T * scale, fp32, causal tile skip.
// grid: (S/128, S/64, B)
// ---------------------------------------------------------------------------
__global__ void __launch_bounds__(256)
scores_gemm(const bf16_t* __restrict__ q, const bf16_t* __restrict__ kk,
            float* __restrict__ sc) {
  const int m0 = blockIdx.x * 128;
  const int n0 = blockIdx.y * 64;
  if (n0 > m0 + 127) return;  // tile entirely above the causal diagonal
  __shared__ bf16_t ldsB[2][64 * LDSROW];
  const int b = blockIdx.z;
  const int tid = threadIdx.x;
  const int lane = tid & 31;
  const int wave = tid >> 5;
  const int mw = m0 + wave * 16;

  v8f acc[4];
  gemm_block(q + ((size_t)b * S_ + mw) * H_, H_,
             kk + ((size_t)b * S_ + n0) * H_, H_, H_, tid,
             &ldsB[0][0], &ldsB[1][0], acc);

  const float scale = 0.03125f;  // 1/sqrt(1024)
  const int hf = lane >> 4, col = lane & 15;
  float* out = sc + (size_t)b * S_ * S_;
#pragma unroll
  for (int t = 0; t < 4; ++t)
#pragma unroll
    for (int j = 0; j < 8; ++j)
      out[(size_t)(mw + hf * 8 + j) * S_ + n0 + t * 16 + col] = acc[t][j] * scale;
}

// ---------------------------------------------------------------------------
// K3: causal softmax, one 256-thread block per (q-row, batch).
//     Writes bf16 probs; zero above the diagonal so K4 needs no mask.
// ---------------------------------------------------------------------------
__global__ void __launch_bounds__(256)
softmax_causal(const float* __restrict__ sc, bf16_t* __restrict__ attn) {
  __shared__ float red[256];
  const int qrow = blockIdx.x;
  const int b = blockIdx.y;
  const int tid = threadIdx.x;
  const float* row = sc + ((size_t)b * S_ + qrow) * S_;
  const int valid = qrow + 1;

  float mx = -3.4e38f;
  for (int kx = tid; kx < valid; kx += 256) mx = fmaxf(mx, row[kx]);
  red[tid] = mx;
  __syncthreads();
  for (int s = 128; s > 0; s >>= 1) {
    if (tid < s) red[tid] = fmaxf(red[tid], red[tid + s]);
    __syncthreads();
  }
  mx = red[0];
  __syncthreads();

  float sum = 0.0f;
  for (int kx = tid; kx < valid; kx += 256) sum += __expf(row[kx] - mx);
  red[tid] = sum;
  __syncthreads();
  for (int s = 128; s > 0; s >>= 1) {
    if (tid < s) red[tid] += red[tid + s];
    __syncthreads();
  }
  const float inv = 1.0f / red[0];

  bf16_t* arow = attn + ((size_t)b * S_ + qrow) * S_;
  for (int kx = tid; kx < S_; kx += 256)
    arow[kx] = (kx < valid) ? (bf16_t)(__expf(row[kx] - mx) * inv) : (bf16_t)0.0f;
}

// ---------------------------------------------------------------------------
// K4: ao = probs @ V, using V^T so the B-operand is K-contiguous.
// grid: (S/128, H/64, B)
// ---------------------------------------------------------------------------
__global__ void __launch_bounds__(256)
av_gemm(const bf16_t* __restrict__ attn, const bf16_t* __restrict__ vT,
        bf16_t* __restrict__ ao) {
  __shared__ bf16_t ldsB[2][64 * LDSROW];
  const int b = blockIdx.z;
  const int tid = threadIdx.x;
  const int lane = tid & 31;
  const int wave = tid >> 5;
  const int m0 = blockIdx.x * 128 + wave * 16;
  const int n0 = blockIdx.y * 64;

  v8f acc[4];
  gemm_block(attn + ((size_t)b * S_ + m0) * S_, S_,
             vT + ((size_t)b * H_ + n0) * S_, S_, S_, tid,
             &ldsB[0][0], &ldsB[1][0], acc);

  const int hf = lane >> 4, col = lane & 15;
#pragma unroll
  for (int t = 0; t < 4; ++t)
#pragma unroll
    for (int j = 0; j < 8; ++j) {
      const int m = m0 + hf * 8 + j;
      ao[((size_t)b * S_ + m) * H_ + n0 + t * 16 + col] = (bf16_t)acc[t][j];
    }
}

// ---------------------------------------------------------------------------
// K5: out = ao @ w_out^T + b_out, fp32 output.
// grid: (B*S/128, H/64)
// ---------------------------------------------------------------------------
__global__ void __launch_bounds__(256)
out_gemm(const bf16_t* __restrict__ ao, const bf16_t* __restrict__ wb,
         const float* __restrict__ bias, float* __restrict__ out) {
  __shared__ bf16_t ldsB[2][64 * LDSROW];
  const int tid = threadIdx.x;
  const int lane = tid & 31;
  const int wave = tid >> 5;
  const int m0 = blockIdx.x * 128 + wave * 16;
  const int n0 = blockIdx.y * 64;

  v8f acc[4];
  gemm_block(ao + (size_t)m0 * H_, H_, wb + (size_t)n0 * H_, H_, H_, tid,
             &ldsB[0][0], &ldsB[1][0], acc);

  const int hf = lane >> 4, col = lane & 15;
#pragma unroll
  for (int t = 0; t < 4; ++t) {
    const int n = n0 + t * 16 + col;
    const float bv = bias[n];
#pragma unroll
    for (int j = 0; j < 8; ++j)
      out[(size_t)(m0 + hf * 8 + j) * H_ + n] = acc[t][j] + bv;
  }
}

// ---------------------------------------------------------------------------
extern "C" void kernel_launch(void* const* d_in, const int* in_sizes, int n_in,
                              void* d_out, int out_size, void* d_ws, size_t ws_size,
                              hipStream_t stream) {
  const float* x     = (const float*)d_in[0];
  const float* w_qkv = (const float*)d_in[1];
  const float* b_qkv = (const float*)d_in[2];
  const float* w_out = (const float*)d_in[3];
  const float* b_out = (const float*)d_in[4];
  float* out = (float*)d_out;

  char* ws = (char*)d_ws;
  size_t off = 0;
  auto carve = [&](size_t bytes) -> void* {
    void* p = (void*)(ws + off);
    off += (bytes + 255) & ~(size_t)255;
    return p;
  };

  const size_t NX = (size_t)B_ * S_ * H_;          // 8,388,608
  bf16_t* xb    = (bf16_t*)carve(NX * 2);
  bf16_t* wqkvb = (bf16_t*)carve((size_t)3 * H_ * H_ * 2);
  bf16_t* woutb = (bf16_t*)carve((size_t)H_ * H_ * 2);
  bf16_t* qb    = (bf16_t*)carve(NX * 2);
  bf16_t* kbuf  = (bf16_t*)carve(NX * 2);
  bf16_t* vTb   = (bf16_t*)carve(NX * 2);
  float*  sc    = (float*)carve((size_t)B_ * S_ * S_ * 4);
  bf16_t* attn  = (bf16_t*)carve((size_t)B_ * S_ * S_ * 2);
  bf16_t* ao    = (bf16_t*)carve(NX * 2);
  (void)ws_size; (void)in_sizes; (void)n_in; (void)out_size;

  // K0: converts
  {
    int n4 = (int)(NX / 4);
    cvt_f32_to_bf16<<<dim3((n4 + 255) / 256), dim3(256), 0, stream>>>(x, xb, n4);
    n4 = 3 * H_ * H_ / 4;
    cvt_f32_to_bf16<<<dim3((n4 + 255) / 256), dim3(256), 0, stream>>>(w_qkv, wqkvb, n4);
    n4 = H_ * H_ / 4;
    cvt_f32_to_bf16<<<dim3((n4 + 255) / 256), dim3(256), 0, stream>>>(w_out, woutb, n4);
  }

  // K1: qkv projection   (M=8192, N=3072, K=1024)
  qkv_gemm<<<dim3((B_ * S_) / 128, (3 * H_) / 64), dim3(256), 0, stream>>>(
      xb, wqkvb, b_qkv, qb, kbuf, vTb);

  // K2: scores           (per batch: M=S, N=S, K=H), causal tile skip
  scores_gemm<<<dim3(S_ / 128, S_ / 64, B_), dim3(256), 0, stream>>>(qb, kbuf, sc);

  // K3: causal softmax
  softmax_causal<<<dim3(S_, B_), dim3(256), 0, stream>>>(sc, attn);

  // K4: probs @ V        (per batch: M=S, N=H, K=S)
  av_gemm<<<dim3(S_ / 128, H_ / 64, B_), dim3(256), 0, stream>>>(attn, vTb, ao);

  // K5: output projection (M=8192, N=1024, K=1024)
  out_gemm<<<dim3((B_ * S_) / 128, H_ / 64), dim3(256), 0, stream>>>(
      ao, woutb, b_out, out);
}
